// Coref_34007551049848
// MI455X (gfx1250) — matrix operand order repaired and checked by reading, not compiled, not run on hardware
//
#include <hip/hip_runtime.h>
#include <hip/hip_bf16.h>

// ---------------- problem constants ----------------
#define TT     768      // tokens
#define EMBD   300
#define EMBP   320      // padded K for embeds
#define NL     800      // 4*LSTM gate width
#define LH     200      // lstm hidden
#define STATE  400
#define STATEP 416
#define HIDP   160      // 150 -> 160
#define SPANW  1101
#define SPANP  1120
#define NS     7635     // total spans widths 1..10
#define NSP    7648     // padded to 16
#define SORTN  8192
#define KSEL   307      // round(768*0.4)
#define NPAIR  46971    // K*(K-1)/2
#define NPAIRP 46976
#define PAIRKP 3360     // 3 segments of 1120

typedef __attribute__((ext_vector_type(16))) __bf16 bf16x16;
typedef __attribute__((ext_vector_type(8)))  float  f32x8;

union FragBF { bf16x16 v; uint4 q[2]; };
union FragF  { f32x8  v; float f[8]; };

// A fragment: 16-bit A 16x32 layout. lanes<16 hold K {0..7,16..23}, lanes>=16 {8..15,24..31}.
__device__ inline FragBF load_fragA(const __hip_bfloat16* __restrict__ Arow, int kc, int half) {
  FragBF f;
  f.q[0] = *(const uint4*)(Arow + kc + half * 8);
  f.q[1] = *(const uint4*)(Arow + kc + 16 + half * 8);
  return f;
}
// B fragment: 16-bit B 32x16 layout. lanes<16 hold K {0..7,16..23} of col N=lane, lanes>=16 {8..15,24..31}.
__device__ inline FragBF load_fragB(const __hip_bfloat16* __restrict__ Brow, int kc, int half) {
  FragBF f;
  f.q[0] = *(const uint4*)(Brow + kc + half * 8);
  f.q[1] = *(const uint4*)(Brow + kc + 16 + half * 8);
  return f;
}
// packed bf16 elementwise product of two fragments (v_pk_mul_bf16 x8)
__device__ inline FragBF mul_frag(FragBF a, FragBF b) {
  union U { uint4 q[2]; __hip_bfloat162 h2[8]; } ua, ub, uc;
  ua.q[0] = a.q[0]; ua.q[1] = a.q[1];
  ub.q[0] = b.q[0]; ub.q[1] = b.q[1];
#pragma unroll
  for (int e = 0; e < 8; e++) uc.h2[e] = __hmul2(ua.h2[e], ub.h2[e]);
  FragBF r; r.q[0] = uc.q[0]; r.q[1] = uc.q[1];
  return r;
}

#define WMMA_BF16(ACC, FA, FB) \
  (ACC) = __builtin_amdgcn_wmma_f32_16x16x32_bf16(false, (FA), false, (FB), (short)0, (ACC), false, false)

// -------- plain bf16 WMMA GEMM: C[M x Npad] = A[M x Kpad] @ B[Npad x Kpad]^T + bias --------
// One wave computes a 16x32 C tile (two N-tiles sharing the A fragment); software-pipelined.
__global__ __launch_bounds__(256)
void gemm_bf16_kernel(const __hip_bfloat16* __restrict__ A,
                      const __hip_bfloat16* __restrict__ B,
                      const float* __restrict__ bias,
                      float* __restrict__ Cf, __hip_bfloat16* __restrict__ Cbf,
                      int M, int Npad, int Kpad, int ldcf, int relu) {
  const int lane  = threadIdx.x & 31;
  const int wave  = threadIdx.y;
  const int tileM = blockIdx.y * 8 + wave;
  const int m0    = tileM * 16;
  const int Mpad  = (M + 15) & ~15;
  if (m0 >= Mpad) return;                       // uniform per wave

  const int half = lane >> 4;
  const int r    = lane & 15;
  const int n0   = blockIdx.x * 32 + r;         // first N tile col for this lane
  const bool twoN = (blockIdx.x * 32 + 16) < Npad;  // block-uniform

  const __hip_bfloat16* Arow  = A + (size_t)(m0 + r) * Kpad;
  const __hip_bfloat16* Brow0 = B + (size_t)n0 * Kpad;
  const __hip_bfloat16* Brow1 = Brow0 + (size_t)16 * Kpad;

  FragF acc0, acc1;
#pragma unroll
  for (int e = 0; e < 8; e++) { acc0.f[e] = 0.f; acc1.f[e] = 0.f; }

  if (twoN) {
    // one-stage pipelined dual-tile loop (branch-free body)
    FragBF fa  = load_fragA(Arow, 0, half);
    FragBF fb0 = load_fragB(Brow0, 0, half);
    FragBF fb1 = load_fragB(Brow1, 0, half);
    int kc = 0;
    for (; kc + 32 < Kpad; kc += 32) {
      __builtin_prefetch(Arow + kc + 160, 0, 1);
      __builtin_prefetch(Brow0 + kc + 160, 0, 1);
      FragBF na  = load_fragA(Arow, kc + 32, half);
      FragBF nb0 = load_fragB(Brow0, kc + 32, half);
      FragBF nb1 = load_fragB(Brow1, kc + 32, half);
      WMMA_BF16(acc0.v, fa.v, fb0.v);
      WMMA_BF16(acc1.v, fa.v, fb1.v);
      fa = na; fb0 = nb0; fb1 = nb1;
    }
    WMMA_BF16(acc0.v, fa.v, fb0.v);
    WMMA_BF16(acc1.v, fa.v, fb1.v);
  } else {
    FragBF fa  = load_fragA(Arow, 0, half);
    FragBF fb0 = load_fragB(Brow0, 0, half);
    int kc = 0;
    for (; kc + 32 < Kpad; kc += 32) {
      __builtin_prefetch(Arow + kc + 160, 0, 1);
      FragBF na  = load_fragA(Arow, kc + 32, half);
      FragBF nb0 = load_fragB(Brow0, kc + 32, half);
      WMMA_BF16(acc0.v, fa.v, fb0.v);
      fa = na; fb0 = nb0;
    }
    WMMA_BF16(acc0.v, fa.v, fb0.v);
  }

  const float bv0 = bias ? bias[n0] : 0.f;
  const float bv1 = (bias && twoN) ? bias[n0 + 16] : 0.f;
#pragma unroll
  for (int e = 0; e < 8; e++) {
    int m = m0 + e + half * 8;                  // C layout: VGPR e -> M=e (+8 for hi lanes)
    if (m < M) {
      float v0 = acc0.f[e] + bv0;
      if (relu) v0 = fmaxf(v0, 0.f);
      if (Cf)  Cf[(size_t)m * ldcf + n0] = v0;
      if (Cbf) Cbf[(size_t)m * Npad + n0] = __float2bfloat16(v0);
      if (twoN) {
        float v1 = acc1.f[e] + bv1;
        if (relu) v1 = fmaxf(v1, 0.f);
        if (Cf)  Cf[(size_t)m * ldcf + n0 + 16] = v1;
        if (Cbf) Cbf[(size_t)m * Npad + n0 + 16] = __float2bfloat16(v1);
      }
    }
  }
}

// -------- pair WMMA GEMM: A rows are [g_i | g_j | g_i*g_j] synthesized on the fly --------
// Kpad = 3*SPANP, Npad = HIDP (two N tiles always valid), ReLU + bf16 out; pipelined segments.
__global__ __launch_bounds__(256)
void gemm_pair_kernel(const int* __restrict__ pii, const int* __restrict__ pjj,
                      const __hip_bfloat16* __restrict__ gf,
                      const __hip_bfloat16* __restrict__ B,
                      const float* __restrict__ bias,
                      __hip_bfloat16* __restrict__ Cbf, int M) {
  const int lane  = threadIdx.x & 31;
  const int wave  = threadIdx.y;
  const int tileM = blockIdx.y * 8 + wave;
  const int m0    = tileM * 16;
  const int Mpad  = (M + 15) & ~15;
  if (m0 >= Mpad) return;

  const int half = lane >> 4;
  const int r    = lane & 15;
  const int n0   = blockIdx.x * 32 + r;

  const int rowA = m0 + r;
  const __hip_bfloat16* gi = gf + (size_t)pii[rowA] * SPANP;
  const __hip_bfloat16* gj = gf + (size_t)pjj[rowA] * SPANP;
  const __hip_bfloat16* Brow0 = B + (size_t)n0 * PAIRKP;
  const __hip_bfloat16* Brow1 = Brow0 + (size_t)16 * PAIRKP;

  FragF acc0, acc1;
#pragma unroll
  for (int e = 0; e < 8; e++) { acc0.f[e] = 0.f; acc1.f[e] = 0.f; }

  // segment 0: A = g_i
  {
    FragBF fa  = load_fragA(gi, 0, half);
    FragBF fb0 = load_fragB(Brow0, 0, half);
    FragBF fb1 = load_fragB(Brow1, 0, half);
    int kc = 0;
    for (; kc + 32 < SPANP; kc += 32) {
      __builtin_prefetch(Brow0 + kc + 160, 0, 1);
      FragBF na  = load_fragA(gi, kc + 32, half);
      FragBF nb0 = load_fragB(Brow0, kc + 32, half);
      FragBF nb1 = load_fragB(Brow1, kc + 32, half);
      WMMA_BF16(acc0.v, fa.v, fb0.v);
      WMMA_BF16(acc1.v, fa.v, fb1.v);
      fa = na; fb0 = nb0; fb1 = nb1;
    }
    WMMA_BF16(acc0.v, fa.v, fb0.v);
    WMMA_BF16(acc1.v, fa.v, fb1.v);
  }
  // segment 1: A = g_j
  {
    FragBF fa  = load_fragA(gj, 0, half);
    FragBF fb0 = load_fragB(Brow0, SPANP, half);
    FragBF fb1 = load_fragB(Brow1, SPANP, half);
    int kc = 0;
    for (; kc + 32 < SPANP; kc += 32) {
      __builtin_prefetch(Brow0 + SPANP + kc + 160, 0, 1);
      FragBF na  = load_fragA(gj, kc + 32, half);
      FragBF nb0 = load_fragB(Brow0, SPANP + kc + 32, half);
      FragBF nb1 = load_fragB(Brow1, SPANP + kc + 32, half);
      WMMA_BF16(acc0.v, fa.v, fb0.v);
      WMMA_BF16(acc1.v, fa.v, fb1.v);
      fa = na; fb0 = nb0; fb1 = nb1;
    }
    WMMA_BF16(acc0.v, fa.v, fb0.v);
    WMMA_BF16(acc1.v, fa.v, fb1.v);
  }
  // segment 2: A = g_i * g_j (packed bf16 product, v_pk_mul_bf16)
  {
    FragBF fa  = mul_frag(load_fragA(gi, 0, half), load_fragA(gj, 0, half));
    FragBF fb0 = load_fragB(Brow0, 2 * SPANP, half);
    FragBF fb1 = load_fragB(Brow1, 2 * SPANP, half);
    int kc = 0;
    for (; kc + 32 < SPANP; kc += 32) {
      __builtin_prefetch(Brow0 + 2 * SPANP + kc + 160, 0, 1);
      FragBF na  = mul_frag(load_fragA(gi, kc + 32, half), load_fragA(gj, kc + 32, half));
      FragBF nb0 = load_fragB(Brow0, 2 * SPANP + kc + 32, half);
      FragBF nb1 = load_fragB(Brow1, 2 * SPANP + kc + 32, half);
      WMMA_BF16(acc0.v, fa.v, fb0.v);
      WMMA_BF16(acc1.v, fa.v, fb1.v);
      fa = na; fb0 = nb0; fb1 = nb1;
    }
    WMMA_BF16(acc0.v, fa.v, fb0.v);
    WMMA_BF16(acc1.v, fa.v, fb1.v);
  }

  const float bv0 = bias[n0];
  const float bv1 = bias[n0 + 16];
#pragma unroll
  for (int e = 0; e < 8; e++) {
    int m = m0 + e + half * 8;
    if (m < M) {
      float v0 = fmaxf(acc0.f[e] + bv0, 0.f);
      float v1 = fmaxf(acc1.f[e] + bv1, 0.f);
      Cbf[(size_t)m * HIDP + n0] = __float2bfloat16(v0);
      Cbf[(size_t)m * HIDP + n0 + 16] = __float2bfloat16(v1);
    }
  }
}

// -------- embedding gather + bf16 pad --------
__global__ void embed_kernel(const int* __restrict__ tokens, const float* __restrict__ emb,
                             float* __restrict__ xf, __hip_bfloat16* __restrict__ xbf) {
  int t = blockIdx.x;
  int tok = tokens[t];
  const float* er = emb + (size_t)tok * EMBD;
  for (int c = threadIdx.x; c < EMBP; c += blockDim.x) {
    float v = (c < EMBD) ? er[c] : 0.f;
    if (c < EMBD) xf[t * EMBD + c] = v;
    xbf[t * EMBP + c] = __float2bfloat16(v);
  }
}

// -------- weight f32 -> bf16 with (segmented) K padding --------
__global__ void convw_kernel(__hip_bfloat16* __restrict__ dst, const float* __restrict__ src,
                             int Nreal, int Npad, int nseg, int segReal, int segPad) {
  int Kpad = nseg * segPad;
  int i = blockIdx.x * blockDim.x + threadIdx.x;
  if (i >= Npad * Kpad) return;
  int n = i / Kpad, k = i % Kpad;
  int s = k / segPad, kk = k % segPad;
  float v = 0.f;
  if (n < Nreal && kk < segReal) v = src[(size_t)n * (nseg * segReal) + s * segReal + kk];
  dst[i] = __float2bfloat16(v);
}

__global__ void padbias_kernel(float* __restrict__ dst, const float* __restrict__ src,
                               int Nreal, int Npad) {
  int i = blockIdx.x * blockDim.x + threadIdx.x;
  if (i >= Npad) return;
  dst[i] = (i < Nreal) ? src[i] : 0.f;
}

// -------- sequential LSTM (one block per direction, h/c in LDS) --------
__global__ __launch_bounds__(NL)
void lstm_kernel(const float* __restrict__ Pf, const float* __restrict__ Pb,
                 const float* __restrict__ WhhF, const float* __restrict__ WhhB,
                 float* __restrict__ h) {
  __shared__ float hprev[LH], cst[LH], z[NL];
  const int dir = blockIdx.x;
  const float* P = dir ? Pb : Pf;
  const float* W = dir ? WhhB : WhhF;
  const int off = dir ? LH : 0;
  int j = threadIdx.x;
  if (j < LH) { hprev[j] = 0.f; cst[j] = 0.f; }
  __syncthreads();
  for (int s = 0; s < TT; s++) {
    int t = dir ? (TT - 1 - s) : s;
    float acc = P[t * NL + j];                  // x@Wih.T + b already folded in
    const float* wr = W + j * LH;
    for (int k = 0; k < LH; k++) acc += hprev[k] * wr[k];
    z[j] = acc;
    __syncthreads();
    if (j < LH) {
      float ig = 1.f / (1.f + expf(-z[j]));
      float fg = 1.f / (1.f + expf(-z[LH + j]));
      float gg = tanhf(z[2 * LH + j]);
      float og = 1.f / (1.f + expf(-z[3 * LH + j]));
      float c = fg * cst[j] + ig * gg;
      cst[j] = c;
      float hh = og * tanhf(c);
      hprev[j] = hh;
      h[t * STATE + off + j] = hh;
    }
    __syncthreads();
  }
}

__global__ void hbf_kernel(const float* __restrict__ h, __hip_bfloat16* __restrict__ hbf) {
  int i = blockIdx.x * blockDim.x + threadIdx.x;
  if (i >= TT * STATEP) return;
  int t = i / STATEP, c = i % STATEP;
  hbf[i] = __float2bfloat16(c < STATE ? h[t * STATE + c] : 0.f);
}

// -------- span feature builder: g = [h[a], h[b], attn-pooled x, n] padded to 1120 bf16 --------
__global__ void gbuild_kernel(const float* __restrict__ h, const float* __restrict__ x,
                              const float* __restrict__ attnC,
                              __hip_bfloat16* __restrict__ gbf,
                              int* __restrict__ i1, int* __restrict__ i2) {
  int s = blockIdx.x;
  __hip_bfloat16* grow = gbf + (size_t)s * SPANP;
  int tid = threadIdx.x;
  if (s >= NS) {
    for (int c = tid; c < SPANP; c += blockDim.x) grow[c] = __float2bfloat16(0.f);
    if (tid == 0) { i1[s] = 0; i2[s] = 0; }
    return;
  }
  int rem = s, n = 1;
  while (rem >= (TT - n + 1)) { rem -= (TT - n + 1); n++; }
  int a = rem, b = a + n - 1;
  if (tid == 0) { i1[s] = a; i2[s] = b; }
  float w[10];
  float mx = -1e30f;
  for (int q = 0; q < n; q++) { w[q] = attnC[(a + q) * 16]; mx = fmaxf(mx, w[q]); }
  float sum = 0.f;
  for (int q = 0; q < n; q++) { w[q] = expf(w[q] - mx); sum += w[q]; }
  float inv = 1.f / sum;
  for (int c = tid; c < SPANP; c += blockDim.x) {
    float v;
    if (c < STATE)            v = h[a * STATE + c];
    else if (c < 2 * STATE)   v = h[b * STATE + (c - STATE)];
    else if (c < 2 * STATE + EMBD) {
      int e = c - 2 * STATE;
      float acc = 0.f;
      for (int q = 0; q < n; q++) acc += w[q] * x[(a + q) * EMBD + e];
      v = acc * inv;
    } else if (c == SPANW - 1) v = (float)n;
    else                       v = 0.f;
    grow[c] = __float2bfloat16(v);
  }
}

// -------- sorting (global-memory bitonic, stable desc via (score desc, idx asc)) --------
__global__ void sortfill_kernel(const float* __restrict__ smC,
                                float* __restrict__ key, int* __restrict__ idx) {
  int i = blockIdx.x * blockDim.x + threadIdx.x;
  if (i >= SORTN) return;
  key[i] = (i < NS) ? smC[(size_t)i * 16] : -3.4e38f;
  idx[i] = i;
}

__global__ void bitonic_kernel(float* __restrict__ key, int* __restrict__ idx, int j, int k) {
  int i = blockIdx.x * blockDim.x + threadIdx.x;
  if (i >= SORTN) return;
  int ixj = i ^ j;
  if (ixj <= i || ixj >= SORTN) return;
  float ka = key[i], kb = key[ixj];
  int ia = idx[i], ib = idx[ixj];
  bool before = (ka > kb) || (ka == kb && ia < ib);
  bool up = ((i & k) == 0);
  bool sw = up ? !before : before;
  if (sw) { key[i] = kb; key[ixj] = ka; idx[i] = ib; idx[ixj] = ia; }
}

// -------- greedy overlap filter (inherently serial) --------
__global__ void filter_kernel(const int* __restrict__ sidx,
                              const int* __restrict__ i1, const int* __restrict__ i2,
                              int* __restrict__ selA, int* __restrict__ selB,
                              int* __restrict__ selID) {
  if (threadIdx.x != 0 || blockIdx.x != 0) return;
  bool taken[TT];
  for (int t = 0; t < TT; t++) taken[t] = false;
  int cnt = 0;
  for (int k = 0; k < NS && cnt < KSEL; k++) {
    int sid = sidx[k];
    int a = i1[sid], b = i2[sid];
    bool all_t = true;
    for (int t = a; t <= b; t++) if (!taken[t]) { all_t = false; break; }
    bool ok = all_t || (!taken[a] && !taken[b]);
    if (ok) {
      selA[cnt] = a; selB[cnt] = b; selID[cnt] = sid; cnt++;
      for (int t = a; t <= b; t++) taken[t] = true;
    }
  }
  for (; cnt < KSEL; cnt++) { selA[cnt] = 0; selB[cnt] = 0; selID[cnt] = 0; }
}

// -------- stable re-sort of K selected spans by (start, score-rank); emit i1_f/i2_f --------
__global__ __launch_bounds__(512)
void selsort_kernel(const int* __restrict__ selA, const int* __restrict__ selB,
                    const int* __restrict__ selID, const float* __restrict__ smC,
                    float* __restrict__ out, float* __restrict__ fsm,
                    int* __restrict__ fsid) {
  __shared__ int key[512], pay[512];
  int t = threadIdx.x;
  if (t < KSEL) { key[t] = selA[t] * 512 + t; pay[t] = t; }
  else          { key[t] = 0x7FFFFFFF;        pay[t] = t; }
  __syncthreads();
  for (int k = 2; k <= 512; k <<= 1)
    for (int j = k >> 1; j > 0; j >>= 1) {
      int ixj = t ^ j;
      if (ixj > t) {
        bool up = ((t & k) == 0);
        int ka = key[t], kb = key[ixj];
        bool sw = up ? (ka > kb) : (ka < kb);
        if (sw) { key[t] = kb; key[ixj] = ka; int p = pay[t]; pay[t] = pay[ixj]; pay[ixj] = p; }
      }
      __syncthreads();
    }
  if (t < KSEL) {
    int r = pay[t];
    out[KSEL * KSEL + t]        = (float)selA[r];
    out[KSEL * KSEL + KSEL + t] = (float)selB[r];
    fsm[t]  = smC[(size_t)selID[r] * 16];
    fsid[t] = selID[r];
  }
}

__global__ void gathergf_kernel(const __hip_bfloat16* __restrict__ gbf,
                                const int* __restrict__ fsid,
                                __hip_bfloat16* __restrict__ gfb) {
  int k = blockIdx.x;
  const __hip_bfloat16* srow = gbf + (size_t)fsid[k] * SPANP;
  __hip_bfloat16* drow = gfb + (size_t)k * SPANP;
  for (int c = threadIdx.x; c < SPANP; c += blockDim.x) drow[c] = srow[c];
}

// -------- tril pair index tables (row-major (i, j<i)) --------
__global__ void pairsidx_kernel(int* __restrict__ pii, int* __restrict__ pjj) {
  int p = blockIdx.x * blockDim.x + threadIdx.x;
  if (p >= NPAIRP) return;
  if (p >= NPAIR) { pii[p] = 0; pjj[p] = 0; return; }
  int i = (int)((1.0 + sqrt(1.0 + 8.0 * (double)p)) * 0.5);
  while ((i * (i - 1)) / 2 > p) i--;
  while (((i + 1) * i) / 2 <= p) i++;
  pii[p] = i;
  pjj[p] = p - (i * (i - 1)) / 2;
}

__global__ void zerof_kernel(float* __restrict__ p, int n) {
  int i = blockIdx.x * blockDim.x + threadIdx.x;
  if (i < n) p[i] = 0.f;
}

__global__ void scatter_kernel(const int* __restrict__ pii, const int* __restrict__ pjj,
                               const float* __restrict__ fsm, const float* __restrict__ saC,
                               float* __restrict__ out) {
  int p = blockIdx.x * blockDim.x + threadIdx.x;
  if (p >= NPAIR) return;
  int i = pii[p], j = pjj[p];
  out[i * KSEL + j] = fsm[i] + fsm[j] + saC[(size_t)p * 16];
}

// ================= host side =================
static inline void launch_gemm(hipStream_t s, const __hip_bfloat16* A,
                               const __hip_bfloat16* B, const float* bias,
                               float* Cf, __hip_bfloat16* Cbf,
                               int M, int Npad, int Kpad, int ldcf, int relu) {
  int Mpad = (M + 15) & ~15;
  dim3 block(32, 8);
  dim3 grid((Npad + 31) / 32, (Mpad / 16 + 7) / 8);
  gemm_bf16_kernel<<<grid, block, 0, s>>>(A, B, bias, Cf, Cbf, M, Npad, Kpad, ldcf, relu);
}

extern "C" void kernel_launch(void* const* d_in, const int* in_sizes, int n_in,
                              void* d_out, int out_size, void* d_ws, size_t ws_size,
                              hipStream_t stream) {
  (void)in_sizes; (void)n_in; (void)out_size; (void)ws_size;
  const int*   tokens = (const int*)  d_in[0];
  const float* emb    = (const float*)d_in[1];
  const float* WihF   = (const float*)d_in[2];
  const float* WhhF   = (const float*)d_in[3];
  const float* bF     = (const float*)d_in[4];
  const float* WihB   = (const float*)d_in[5];
  const float* WhhB   = (const float*)d_in[6];
  const float* bB     = (const float*)d_in[7];
  const float* aW1 = (const float*)d_in[8];  const float* ab1 = (const float*)d_in[9];
  const float* aW2 = (const float*)d_in[10]; const float* ab2 = (const float*)d_in[11];
  const float* aW3 = (const float*)d_in[12]; const float* ab3 = (const float*)d_in[13];
  const float* sW1 = (const float*)d_in[14]; const float* sb1 = (const float*)d_in[15];
  const float* sW2 = (const float*)d_in[16]; const float* sb2 = (const float*)d_in[17];
  const float* sW3 = (const float*)d_in[18]; const float* sb3 = (const float*)d_in[19];
  const float* pW1 = (const float*)d_in[20]; const float* pb1 = (const float*)d_in[21];
  const float* pW2 = (const float*)d_in[22]; const float* pb2 = (const float*)d_in[23];
  const float* pW3 = (const float*)d_in[24]; const float* pb3 = (const float*)d_in[25];
  float* out = (float*)d_out;

  // bump allocator over workspace (256B aligned slabs)
  char* base = (char*)d_ws;
  size_t off = 0;
  auto alloc = [&](size_t bytes) -> char* {
    char* p = base + off;
    off += (bytes + 255) & ~(size_t)255;
    return p;
  };
  typedef __hip_bfloat16 bf;
  float* xf   = (float*)alloc((size_t)TT * EMBD * 4);
  bf* xbf     = (bf*)   alloc((size_t)TT * EMBP * 2);
  bf* WihFb   = (bf*)   alloc((size_t)NL * EMBP * 2);
  bf* WihBb   = (bf*)   alloc((size_t)NL * EMBP * 2);
  float* Pf   = (float*)alloc((size_t)TT * NL * 4);
  float* Pb   = (float*)alloc((size_t)TT * NL * 4);
  float* hbuf = (float*)alloc((size_t)TT * STATE * 4);
  bf* hbfb    = (bf*)   alloc((size_t)TT * STATEP * 2);
  bf* aW1b = (bf*)alloc((size_t)HIDP * STATEP * 2);
  bf* aW2b = (bf*)alloc((size_t)HIDP * HIDP * 2);
  bf* aW3b = (bf*)alloc((size_t)16 * HIDP * 2);
  float* ab1p = (float*)alloc(HIDP * 4);
  float* ab2p = (float*)alloc(HIDP * 4);
  float* ab3p = (float*)alloc(16 * 4);
  bf* A1 = (bf*)alloc((size_t)TT * HIDP * 2);
  bf* A2 = (bf*)alloc((size_t)TT * HIDP * 2);
  float* attnC = (float*)alloc((size_t)TT * 16 * 4);
  bf* sW1b = (bf*)alloc((size_t)HIDP * SPANP * 2);
  bf* sW2b = (bf*)alloc((size_t)HIDP * HIDP * 2);
  bf* sW3b = (bf*)alloc((size_t)16 * HIDP * 2);
  float* sb1p = (float*)alloc(HIDP * 4);
  float* sb2p = (float*)alloc(HIDP * 4);
  float* sb3p = (float*)alloc(16 * 4);
  bf* gbf = (bf*)alloc((size_t)NSP * SPANP * 2);
  int* i1a = (int*)alloc((size_t)NSP * 4);
  int* i2a = (int*)alloc((size_t)NSP * 4);
  bf* S1 = (bf*)alloc((size_t)NSP * HIDP * 2);
  bf* S2 = (bf*)alloc((size_t)NSP * HIDP * 2);
  float* smC = (float*)alloc((size_t)NSP * 16 * 4);
  float* skey = (float*)alloc((size_t)SORTN * 4);
  int*   sidx = (int*)  alloc((size_t)SORTN * 4);
  int* selA  = (int*)alloc((size_t)KSEL * 4);
  int* selB  = (int*)alloc((size_t)KSEL * 4);
  int* selID = (int*)alloc((size_t)KSEL * 4);
  float* fsm = (float*)alloc((size_t)KSEL * 4);
  int* fsid  = (int*)  alloc((size_t)KSEL * 4);
  bf* gfb = (bf*)alloc((size_t)KSEL * SPANP * 2);
  int* pii = (int*)alloc((size_t)NPAIRP * 4);
  int* pjj = (int*)alloc((size_t)NPAIRP * 4);
  bf* P1 = (bf*)alloc((size_t)NPAIRP * HIDP * 2);
  bf* P2 = (bf*)alloc((size_t)NPAIRP * HIDP * 2);
  bf* pW1b = (bf*)alloc((size_t)HIDP * PAIRKP * 2);
  bf* pW2b = (bf*)alloc((size_t)HIDP * HIDP * 2);
  bf* pW3b = (bf*)alloc((size_t)16 * HIDP * 2);
  float* pb1p = (float*)alloc(HIDP * 4);
  float* pb2p = (float*)alloc(HIDP * 4);
  float* pb3p = (float*)alloc(16 * 4);
  float* saC = (float*)alloc((size_t)NPAIRP * 16 * 4);

  auto grd = [](long n, int b) { return (int)((n + b - 1) / b); };

  // ---- weight / bias conversions (bf16, zero-padded) ----
  convw_kernel<<<grd((long)NL * EMBP, 256), 256, 0, stream>>>(WihFb, WihF, NL, NL, 1, EMBD, EMBP);
  convw_kernel<<<grd((long)NL * EMBP, 256), 256, 0, stream>>>(WihBb, WihB, NL, NL, 1, EMBD, EMBP);
  convw_kernel<<<grd((long)HIDP * STATEP, 256), 256, 0, stream>>>(aW1b, aW1, 150, HIDP, 1, STATE, STATEP);
  convw_kernel<<<grd((long)HIDP * HIDP, 256), 256, 0, stream>>>(aW2b, aW2, 150, HIDP, 1, 150, HIDP);
  convw_kernel<<<grd((long)16 * HIDP, 256), 256, 0, stream>>>(aW3b, aW3, 1, 16, 1, 150, HIDP);
  convw_kernel<<<grd((long)HIDP * SPANP, 256), 256, 0, stream>>>(sW1b, sW1, 150, HIDP, 1, SPANW, SPANP);
  convw_kernel<<<grd((long)HIDP * HIDP, 256), 256, 0, stream>>>(sW2b, sW2, 150, HIDP, 1, 150, HIDP);
  convw_kernel<<<grd((long)16 * HIDP, 256), 256, 0, stream>>>(sW3b, sW3, 1, 16, 1, 150, HIDP);
  convw_kernel<<<grd((long)HIDP * PAIRKP, 256), 256, 0, stream>>>(pW1b, pW1, 150, HIDP, 3, SPANW, SPANP);
  convw_kernel<<<grd((long)HIDP * HIDP, 256), 256, 0, stream>>>(pW2b, pW2, 150, HIDP, 1, 150, HIDP);
  convw_kernel<<<grd((long)16 * HIDP, 256), 256, 0, stream>>>(pW3b, pW3, 1, 16, 1, 150, HIDP);
  padbias_kernel<<<1, 256, 0, stream>>>(ab1p, ab1, 150, HIDP);
  padbias_kernel<<<1, 256, 0, stream>>>(ab2p, ab2, 150, HIDP);
  padbias_kernel<<<1, 256, 0, stream>>>(ab3p, ab3, 1, 16);
  padbias_kernel<<<1, 256, 0, stream>>>(sb1p, sb1, 150, HIDP);
  padbias_kernel<<<1, 256, 0, stream>>>(sb2p, sb2, 150, HIDP);
  padbias_kernel<<<1, 256, 0, stream>>>(sb3p, sb3, 1, 16);
  padbias_kernel<<<1, 256, 0, stream>>>(pb1p, pb1, 150, HIDP);
  padbias_kernel<<<1, 256, 0, stream>>>(pb2p, pb2, 150, HIDP);
  padbias_kernel<<<1, 256, 0, stream>>>(pb3p, pb3, 1, 16);

  // ---- embeds, LSTM input projection (WMMA), recurrence ----
  embed_kernel<<<TT, 128, 0, stream>>>(tokens, emb, xf, xbf);
  launch_gemm(stream, xbf, WihFb, bF, Pf, 0, TT, NL, EMBP, NL, 0);
  launch_gemm(stream, xbf, WihBb, bB, Pb, 0, TT, NL, EMBP, NL, 0);
  lstm_kernel<<<2, NL, 0, stream>>>(Pf, Pb, WhhF, WhhB, hbuf);
  hbf_kernel<<<grd((long)TT * STATEP, 256), 256, 0, stream>>>(hbuf, hbfb);

  // ---- attention FFNN (WMMA chain) ----
  launch_gemm(stream, hbfb, aW1b, ab1p, 0, A1, TT, HIDP, STATEP, 0, 1);
  launch_gemm(stream, A1, aW2b, ab2p, 0, A2, TT, HIDP, HIDP, 0, 1);
  launch_gemm(stream, A2, aW3b, ab3p, attnC, 0, TT, 16, HIDP, 16, 0);

  // ---- span features + span FFNN ----
  gbuild_kernel<<<NSP, 128, 0, stream>>>(hbuf, xf, attnC, gbf, i1a, i2a);
  launch_gemm(stream, gbf, sW1b, sb1p, 0, S1, NS, HIDP, SPANP, 0, 1);
  launch_gemm(stream, S1, sW2b, sb2p, 0, S2, NS, HIDP, HIDP, 0, 1);
  launch_gemm(stream, S2, sW3b, sb3p, smC, 0, NS, 16, HIDP, 16, 0);

  // ---- stable descending sort of span scores ----
  sortfill_kernel<<<SORTN / 256, 256, 0, stream>>>(smC, skey, sidx);
  for (int k = 2; k <= SORTN; k <<= 1)
    for (int j = k >> 1; j > 0; j >>= 1)
      bitonic_kernel<<<SORTN / 256, 256, 0, stream>>>(skey, sidx, j, k);

  // ---- greedy overlap filter, re-sort by start, gather selected ----
  filter_kernel<<<1, 32, 0, stream>>>(sidx, i1a, i2a, selA, selB, selID);
  zerof_kernel<<<grd((long)KSEL * KSEL, 256), 256, 0, stream>>>(out, KSEL * KSEL);
  selsort_kernel<<<1, 512, 0, stream>>>(selA, selB, selID, smC, out, fsm, fsid);
  gathergf_kernel<<<KSEL, 256, 0, stream>>>(gbf, fsid, gfb);

  // ---- pair FFNN (dominant GEMM; A synthesized per segment, no inner-loop division) ----
  pairsidx_kernel<<<grd(NPAIRP, 256), 256, 0, stream>>>(pii, pjj);
  {
    dim3 block(32, 8);
    dim3 grid(HIDP / 32, (NPAIRP / 16 + 7) / 8);
    gemm_pair_kernel<<<grid, block, 0, stream>>>(pii, pjj, gfb, pW1b, pb1p, P1, NPAIR);
  }
  launch_gemm(stream, P1, pW2b, pb2p, 0, P2, NPAIR, HIDP, HIDP, 0, 1);
  launch_gemm(stream, P2, pW3b, pb3p, saC, 0, NPAIR, 16, HIDP, 16, 0);

  // ---- final antecedent score matrix ----
  scatter_kernel<<<grd(NPAIR, 256), 256, 0, stream>>>(pii, pjj, fsm, saC, out);
}
